// LDW_up_52639119180442
// MI455X (gfx1250) — matrix operand order repaired and checked by reading, not compiled, not run on hardware
//
#include <hip/hip_runtime.h>
#include <stdint.h>

// ---- problem constants ----
#define CCH    256
#define HPIX   128
#define WPIX   128
#define PPIX   (HPIX*WPIX)
#define C4N    64
#define TILE_N 64

#define XSTRIDE   528              // bytes per pixel row in LDS (264 bf16, 16B aligned)
#define STAGE_OFF 33792            // after bf16 X region (64*528)
#define STAGE_SZ  4096             // 16 ch x 64 px fp32

#if defined(__AMDGCN__) && __has_builtin(__builtin_amdgcn_global_load_async_to_lds_b64)
#define HAVE_ASYNC 1
#endif

typedef __attribute__((ext_vector_type(16))) __bf16 bf16x16;
typedef __attribute__((ext_vector_type(8)))  float  f32x8;

union FragU { uint4 u[2]; bf16x16 v; };

__device__ __forceinline__ uint16_t f2bf(float f) {
    uint32_t u = __float_as_uint(f);
    return (uint16_t)((u + 0x7FFFu + ((u >> 16) & 1u)) >> 16);   // RNE
}

#ifdef HAVE_ASYNC
typedef int v2i __attribute__((ext_vector_type(2)));
typedef __attribute__((address_space(1))) v2i* gp64;   // global int2*
typedef __attribute__((address_space(3))) v2i* lp64;   // LDS int2*
__device__ __forceinline__ void async_cp_b64(const float* g, const unsigned char* l) {
    __builtin_amdgcn_global_load_async_to_lds_b64(
        (gp64)(uintptr_t)g, (lp64)(uint32_t)(uintptr_t)l, 0, 0);
}
#endif

// ------------------------------------------------------------------
// Prep: conv1x1_w (256x256 f32 [o][c]) -> bf16 WMMA A-fragment layout
// (ISA 7.12.2). Fragment (mt*8+kt): 512 bf16 = 32 lanes x 16 elems.
// ------------------------------------------------------------------
__global__ __launch_bounds__(256)
void prep_w_bf16(const float* __restrict__ Wm, uint16_t* __restrict__ wsA) {
    int gid  = blockIdx.x * 256 + threadIdx.x;
    int frag = gid >> 9;
    int r    = gid & 511;
    int lane = r >> 4;
    int e    = r & 15;
    int mt   = frag >> 3;
    int kt   = frag & 7;
    int m    = mt * 16 + (lane & 15);
    int k;
    if (lane < 16) k = kt * 32 + ((e < 8) ? e     : e + 8);
    else           k = kt * 32 + ((e < 8) ? e + 8 : e + 16);
    wsA[gid] = f2bf(Wm[m * CCH + k]);
}

// ------------------------------------------------------------------
// Fused bf16-WMMA 1x1-conv GEMM + wavelet 2x2 upsample.
// 256 threads (8 waves); each block owns 64 pixels of one h-row.
// Wave w owns m-tiles {2w, 2w+1} x all 4 n-tiles (W read once/block).
// ------------------------------------------------------------------
__global__ __launch_bounds__(256)
void ldw_fused(const float* __restrict__ x,
               const uint16_t* __restrict__ wsA,
               const float* __restrict__ lp_v, const float* __restrict__ hp_v,
               const float* __restrict__ lp_h, const float* __restrict__ hp_h,
               float* __restrict__ out) {
    __shared__ __align__(16) unsigned char smem[65536];
    float* ys = (float*)smem;                     // after GEMM: [256 ch][64 px] f32

    const int t    = threadIdx.x;
    const int blk  = blockIdx.x;                  // 4096
    const int b    = blk >> 8;
    const int tile = blk & 255;
    const int p0   = tile * TILE_N;
    const int h    = p0 >> 7;
    const int w0   = p0 & 127;

    const float* xb = x + (size_t)b * CCH * PPIX;

    // ================= stage X tile to LDS as bf16 [pixel][channel] =================
#ifdef HAVE_ASYNC
    {
        // Async pipeline: 16 chunks of 16 channels. Each lane async-copies
        // (2 channels x 2 pixels) per chunk into a ping-pong fp32 staging
        // buffer, then converts its OWN data from the previous chunk
        // (wave-local dependency -> only s_wait_asynccnt, no barriers).
        int j0 = (t & 31) * 2;                    // pixel pair
        int cp = (t >> 5) * 2;                    // channel pair within chunk
        const float* g0 = xb + p0 + j0;
        unsigned char* stA = smem + STAGE_OFF;

        for (int cc = 0; cc < 16; ++cc) {
            int buf = cc & 1;
            unsigned char* st = stA + buf * STAGE_SZ;   // [16 ch][64 px] f32
            int c0 = cc * 16;
            async_cp_b64(g0 + (size_t)(c0 + cp)     * PPIX, st + (cp * 64 + j0) * 4);
            async_cp_b64(g0 + (size_t)(c0 + cp + 1) * PPIX, st + ((cp + 1) * 64 + j0) * 4);
            if (cc) {
                asm volatile("s_wait_asynccnt 0x2" ::: "memory");
                const unsigned char* sp = stA + (buf ^ 1) * STAGE_SZ;
                float2 v = *(const float2*)(sp + (cp * 64 + j0) * 4);
                float2 w = *(const float2*)(sp + ((cp + 1) * 64 + j0) * 4);
                int c = (cc - 1) * 16 + cp;
                *(uint32_t*)(smem + j0 * XSTRIDE + c * 2) =
                    (uint32_t)f2bf(v.x) | ((uint32_t)f2bf(w.x) << 16);
                *(uint32_t*)(smem + (j0 + 1) * XSTRIDE + c * 2) =
                    (uint32_t)f2bf(v.y) | ((uint32_t)f2bf(w.y) << 16);
            }
        }
        asm volatile("s_wait_asynccnt 0x0" ::: "memory");
        {   // convert final chunk (cc = 15, buf = 1)
            const unsigned char* sp = stA + STAGE_SZ;
            float2 v = *(const float2*)(sp + (cp * 64 + j0) * 4);
            float2 w = *(const float2*)(sp + ((cp + 1) * 64 + j0) * 4);
            int c = 15 * 16 + cp;
            *(uint32_t*)(smem + j0 * XSTRIDE + c * 2) =
                (uint32_t)f2bf(v.x) | ((uint32_t)f2bf(w.x) << 16);
            *(uint32_t*)(smem + (j0 + 1) * XSTRIDE + c * 2) =
                (uint32_t)f2bf(v.y) | ((uint32_t)f2bf(w.y) << 16);
        }
    }
#else
    {
        int j0 = (t & 15) * 4;                    // 4 pixels
        int cg = t >> 4;                          // 0..15
        for (int it = 0; it < 8; ++it) {
            int c = it * 32 + cg * 2;
            float4 v = *(const float4*)(xb + (size_t)c       * PPIX + p0 + j0);
            float4 w = *(const float4*)(xb + (size_t)(c + 1) * PPIX + p0 + j0);
            const float* vp = (const float*)&v;
            const float* wp = (const float*)&w;
            #pragma unroll
            for (int s = 0; s < 4; ++s)
                *(uint32_t*)(smem + (j0 + s) * XSTRIDE + c * 2) =
                    (uint32_t)f2bf(vp[s]) | ((uint32_t)f2bf(wp[s]) << 16);
        }
    }
#endif
    __syncthreads();

    // ================= GEMM: D = W(16x32) x X(32x16), bf16 WMMA =================
    const int wid  = t >> 5;
    const int lane = t & 31;
    const int mt0  = wid * 2;                     // wave owns m-tiles mt0, mt0+1
    const int khalf = (lane >> 4) * 8;

    f32x8 acc[2][4];
    {
        f32x8 z = {0.f, 0.f, 0.f, 0.f, 0.f, 0.f, 0.f, 0.f};
        #pragma unroll
        for (int m = 0; m < 2; ++m)
            #pragma unroll
            for (int nt = 0; nt < 4; ++nt) acc[m][nt] = z;
    }

    const uint16_t* abase = wsA + (size_t)mt0 * 8 * 512 + lane * 16;
    FragU abuf[2][2];                             // double-buffered A fragments

    auto LOADA = [&](int kt, FragU* d) {
        const uint16_t* p = abase + (size_t)kt * 512;
        d[0].u[0] = *(const uint4*)(p);
        d[0].u[1] = *(const uint4*)(p + 8);
        const uint16_t* q = p + 8 * 512;          // m-tile mt0+1
        d[1].u[0] = *(const uint4*)(q);
        d[1].u[1] = *(const uint4*)(q + 8);
    };

    LOADA(0, abuf[0]);
    #pragma unroll
    for (int kt = 0; kt < 8; ++kt) {
        FragU* ac = abuf[kt & 1];
        if (kt < 7) LOADA(kt + 1, abuf[(kt + 1) & 1]);   // prefetch next k-step
        const unsigned char* brow = smem + (lane & 15) * XSTRIDE + (kt * 32 + khalf) * 2;
        #pragma unroll
        for (int nt = 0; nt < 4; ++nt) {
            FragU bu;
            const unsigned char* ba = brow + nt * 16 * XSTRIDE;
            bu.u[0] = *(const uint4*)(ba);
            bu.u[1] = *(const uint4*)(ba + 32);
            acc[0][nt] = __builtin_amdgcn_wmma_f32_16x16x32_bf16(
                false, ac[0].v, false, bu.v, (short)0, acc[0][nt], false, false);
            acc[1][nt] = __builtin_amdgcn_wmma_f32_16x16x32_bf16(
                false, ac[1].v, false, bu.v, (short)0, acc[1][nt], false, false);
        }
    }
    __syncthreads();   // all GEMM LDS reads done before Y overwrites

    // ================= spill Y to LDS f32 [channel][pixel] =================
    {
        int nnl  = lane & 15;
        int mofs = (lane >> 4) * 8;               // C-layout: M = r + 8*(lane>=16)
        #pragma unroll
        for (int m = 0; m < 2; ++m) {
            int obase = (mt0 + m) * 16 + mofs;
            #pragma unroll
            for (int nt = 0; nt < 4; ++nt) {
                int nn = nt * 16 + nnl;
                #pragma unroll
                for (int r = 0; r < 8; ++r)
                    ys[(obase + r) * TILE_N + nn] = acc[m][nt][r];
            }
        }
    }
    __syncthreads();

    // ================= wavelet 2x2 upsample =================
    {
        int nn   = t & 63;
        int c4b  = t >> 6;
        int wcol = w0 + nn;
        float* outb = out + (size_t)b * C4N * 256 * 256;
        const float2* lpv2 = (const float2*)lp_v;
        const float2* hpv2 = (const float2*)hp_v;
        const float2* lph2 = (const float2*)lp_h;
        const float2* hph2 = (const float2*)hp_h;
        #pragma unroll
        for (int i = 0; i < 16; ++i) {
            int c4 = c4b + i * 4;
            float a  = ys[(c4      ) * TILE_N + nn];
            float bq = ys[(c4 +  64) * TILE_N + nn];
            float cq = ys[(c4 + 128) * TILE_N + nn];
            float dq = ys[(c4 + 192) * TILE_N + nn];
            float2 LV = lpv2[c4], HV = hpv2[c4], LH = lph2[c4], HH = hph2[c4];
            float s0 = LV.x * a  + LV.y * bq;     // output row 2h
            float s1 = HV.x * a  + HV.y * bq;     // output row 2h+1
            float t0 = LV.x * cq + LV.y * dq;
            float t1 = HV.x * cq + HV.y * dq;
            float2 r0 = make_float2(LH.x * s0 + LH.y * t0, HH.x * s0 + HH.y * t0);
            float2 r1 = make_float2(LH.x * s1 + LH.y * t1, HH.x * s1 + HH.y * t1);
            size_t o0 = ((size_t)c4 * 256 + 2 * h) * 256 + 2 * wcol;
            *(float2*)(outb + o0)       = r0;
            *(float2*)(outb + o0 + 256) = r1;
        }
    }
}

extern "C" void kernel_launch(void* const* d_in, const int* in_sizes, int n_in,
                              void* d_out, int out_size, void* d_ws, size_t ws_size,
                              hipStream_t stream) {
    const float* x    = (const float*)d_in[0];
    const float* Wm   = (const float*)d_in[1];
    const float* lp_v = (const float*)d_in[2];
    const float* hp_v = (const float*)d_in[3];
    const float* lp_h = (const float*)d_in[4];
    const float* hp_h = (const float*)d_in[5];
    uint16_t* wsA = (uint16_t*)d_ws;      // 16*8 fragments * 512 bf16 = 128 KB

    prep_w_bf16<<<256, 256, 0, stream>>>(Wm, wsA);
    ldw_fused<<<4096, 256, 0, stream>>>(x, wsA, lp_v, hp_v, lp_h, hp_h, (float*)d_out);
}
